// MOELayer_15951508537921
// MI455X (gfx1250) — compile-verified, matrix operand order checked by reading, not run on hardware
//
#include <hip/hip_runtime.h>

// Problem constants (from reference: B=4, S=2048, D=1024, E=8, H=2816, top-2)
constexpr int Tv = 8192;   // B*S tokens
constexpr int Dv = 1024;   // model dim
constexpr int Ev = 8;      // experts
constexpr int Hv = 2816;   // ffn hidden

typedef __bf16 bf16;
typedef __attribute__((ext_vector_type(16))) __bf16 v16bf;
typedef __attribute__((ext_vector_type(8)))  __bf16 v8bf;
typedef __attribute__((ext_vector_type(8)))  float  v8f;
typedef __attribute__((ext_vector_type(4)))  int    v4i;

// LDS tile row stride (32 K values + 8 pad) -> 80B rows, conflict-free b128 frag loads
constexpr int LDSW = 40;

// gfx1250 async global->LDS copy path (ASYNCcnt-tracked), with safe fallback.
#if defined(__has_builtin)
#if __has_builtin(__builtin_amdgcn_global_load_async_to_lds_b128) && \
    __has_builtin(__builtin_amdgcn_s_wait_asynccnt)
#define USE_ASYNC_LDS 1
#endif
#endif
#ifndef USE_ASYNC_LDS
#define USE_ASYNC_LDS 0
#endif

// Load a 16x32 bf16 WMMA fragment from an LDS tile stored [row][k] (row = M for A,
// row = N for B). Layout per CDNA5 ISA 7.12.2:
//   lanes 0-15 : row = rbase+lane,    K = {0..7} then {16..23}
//   lanes 16-31: row = rbase+lane-16, K = {8..15} then {24..31}
__device__ __forceinline__ v16bf ld_frag(const bf16* tile, int rbase, int lane) {
  int r  = rbase + (lane & 15);
  int kh = (lane >> 4) << 3;                    // 0 or 8
  const bf16* p = tile + r * LDSW;
  v8bf lo = *(const v8bf*)(p + kh);             // K = kh .. kh+7
  v8bf hi = *(const v8bf*)(p + 16 + kh);        // K = 16+kh .. 16+kh+7
  v16bf f;
#pragma unroll
  for (int i = 0; i < 8; ++i) { f[i] = lo[i]; f[i + 8] = hi[i]; }
  return f;
}

__device__ __forceinline__ void cvt_store4(bf16* d, float4 v) {
  d[0] = (bf16)v.x; d[1] = (bf16)v.y; d[2] = (bf16)v.z; d[3] = (bf16)v.w;
}

// ---------------------------------------------------------------------------
// Router: one wave per token. logits = x[t] . gate_w[e], top-2 softmax,
// append (token, weight) into per-expert compacted lists.
// ---------------------------------------------------------------------------
__global__ void __launch_bounds__(256)
router_kernel(const float* __restrict__ x, const float* __restrict__ gw,
              int* __restrict__ counts, int* __restrict__ tok_list,
              float* __restrict__ tok_w) {
  const int wave = threadIdx.x >> 5;
  const int lane = threadIdx.x & 31;
  const int t = blockIdx.x * 8 + wave;
  const float* xp = x + (size_t)t * Dv;

  float lg[Ev];
#pragma unroll
  for (int e = 0; e < Ev; ++e) lg[e] = 0.f;

  for (int k = lane; k < Dv; k += 32) {
    float xv = xp[k];
#pragma unroll
    for (int e = 0; e < Ev; ++e) lg[e] += xv * gw[e * Dv + k];
  }
#pragma unroll
  for (int e = 0; e < Ev; ++e) {
#pragma unroll
    for (int off = 16; off > 0; off >>= 1) lg[e] += __shfl_xor(lg[e], off, 32);
  }

  if (lane == 0) {
    float l0 = -3.4e38f; int e0 = 0;
#pragma unroll
    for (int e = 0; e < Ev; ++e) if (lg[e] > l0) { l0 = lg[e]; e0 = e; }
    float l1 = -3.4e38f; int e1 = 0;
#pragma unroll
    for (int e = 0; e < Ev; ++e) if (e != e0 && lg[e] > l1) { l1 = lg[e]; e1 = e; }
    float p  = __expf(l1 - l0);          // softmax over the two kept logits
    float w0 = __builtin_amdgcn_rcpf(1.f + p);
    float w1 = 1.f - w0;
    int s0 = atomicAdd(&counts[e0], 1);
    tok_list[e0 * Tv + s0] = t; tok_w[e0 * Tv + s0] = w0;
    int s1 = atomicAdd(&counts[e1], 1);
    tok_list[e1 * Tv + s1] = t; tok_w[e1 * Tv + s1] = w1;
  }
}

// ---------------------------------------------------------------------------
// h = silu(Xg @ w1[e]^T) * (Xg @ w3[e]^T) over gathered tokens of expert e.
// Tile 128(tokens) x 128(H cols), K = D in chunks of 32, ping-pong LDS so
// chunk k+1 stages (f32 load -> bf16 cvt -> ds_store) while chunk k's WMMAs
// run on sibling waves. 8 waves: 4(M) x 2(N), each wave 32x64.
// ---------------------------------------------------------------------------
__global__ void __launch_bounds__(256)
moe_h_kernel(const float* __restrict__ x, const float* __restrict__ w1,
             const float* __restrict__ w3, const int* __restrict__ counts,
             const int* __restrict__ tok_list, bf16* __restrict__ hb, int e) {
  __shared__ __align__(16) bf16 As[2][128 * LDSW];
  __shared__ __align__(16) bf16 B1s[2][128 * LDSW];
  __shared__ __align__(16) bf16 B3s[2][128 * LDSW];
  __shared__ int toks[128];

  const int cnt = counts[e];
  const int m0 = blockIdx.x * 128;
  if (m0 >= cnt) return;                       // uniform early-exit (sparse dispatch)
  const int n0 = blockIdx.y * 128;
  const int tid = threadIdx.x;

  if (tid < 128) {
    int s = m0 + tid;
    toks[tid] = tok_list[e * Tv + (s < cnt ? s : cnt - 1)];
  }
  __syncthreads();

  const int wv = tid >> 5, lane = tid & 31;
  const int wm = (wv & 3) << 5;                // 0,32,64,96
  const int wn = (wv >> 2) << 6;               // 0,64

  v8f a1[2][4] = {};
  v8f a3[2][4] = {};

  const float* w1p = w1 + ((size_t)e * Hv + n0) * Dv;
  const float* w3p = w3 + ((size_t)e * Hv + n0) * Dv;

  // stage one 128x32 chunk (3 tiles) into buffer `buf`
  auto stage = [&](int buf, int kb) {
    for (int i = tid; i < 1024; i += 256) {
      int r = i >> 3, kg = (i & 7) << 2;
      float4 va = *(const float4*)(x + (size_t)toks[r] * Dv + kb + kg);
      float4 v1 = *(const float4*)(w1p + (size_t)r * Dv + kb + kg);
      float4 v3 = *(const float4*)(w3p + (size_t)r * Dv + kb + kg);
      cvt_store4(&As[buf][r * LDSW + kg], va);
      cvt_store4(&B1s[buf][r * LDSW + kg], v1);
      cvt_store4(&B3s[buf][r * LDSW + kg], v3);
    }
  };

  stage(0, 0);
  __syncthreads();

  constexpr int NCH = Dv / 32;                 // 32 chunks
  for (int kc = 0; kc < NCH; ++kc) {
    const int cur = kc & 1;
    if (kc + 1 < NCH) stage(cur ^ 1, (kc + 1) * 32);   // overlaps with WMMA below
#pragma unroll
    for (int mi = 0; mi < 2; ++mi) {
      v16bf af = ld_frag(As[cur], wm + mi * 16, lane);
#pragma unroll
      for (int ni = 0; ni < 4; ++ni) {
        v16bf b1 = ld_frag(B1s[cur], wn + ni * 16, lane);
        v16bf b3 = ld_frag(B3s[cur], wn + ni * 16, lane);
        a1[mi][ni] = __builtin_amdgcn_wmma_f32_16x16x32_bf16(
            false, af, false, b1, (short)0, a1[mi][ni], false, false);
        a3[mi][ni] = __builtin_amdgcn_wmma_f32_16x16x32_bf16(
            false, af, false, b3, (short)0, a3[mi][ni], false, false);
      }
    }
    __syncthreads();
  }

  // fused SwiGLU epilogue (fast rcp-based silu), bf16 store per slot
  const int hi8 = (lane >> 4) << 3, l15 = lane & 15;
#pragma unroll
  for (int mi = 0; mi < 2; ++mi)
#pragma unroll
    for (int ni = 0; ni < 4; ++ni)
#pragma unroll
      for (int i = 0; i < 8; ++i) {
        int r = wm + mi * 16 + hi8 + i;
        int slot = m0 + r;
        if (slot < cnt) {
          int col = n0 + wn + ni * 16 + l15;
          float a = a1[mi][ni][i], b = a3[mi][ni][i];
          float h = a * __builtin_amdgcn_rcpf(1.f + __expf(-a)) * b;  // silu(a)*b
          hb[(size_t)slot * Hv + col] = (bf16)h;
        }
      }
}

// ---------------------------------------------------------------------------
// out[token] += cw * (h @ w2[e]^T). Tile 128(slots) x 128(D cols), K = H.
// A-tile (hb, already bf16) is staged with async global->LDS copies
// (ASYNCcnt); B-tile (w2 f32) goes through the cvt register path.
// Experts execute serially on the stream, so plain RMW on `out` is safe.
// ---------------------------------------------------------------------------
__global__ void __launch_bounds__(256)
moe_o_kernel(const bf16* __restrict__ hb, const float* __restrict__ w2,
             const int* __restrict__ counts, const int* __restrict__ tok_list,
             const float* __restrict__ tok_w, float* __restrict__ out, int e) {
  __shared__ __align__(16) bf16 As[2][128 * LDSW];
  __shared__ __align__(16) bf16 Bs[2][128 * LDSW];
  __shared__ int toks[128];
  __shared__ float wts[128];

  const int cnt = counts[e];
  const int m0 = blockIdx.x * 128;
  if (m0 >= cnt) return;
  const int n0 = blockIdx.y * 128;
  const int tid = threadIdx.x;

  if (tid < 128) {
    int s = m0 + tid;
    int sc = s < cnt ? s : cnt - 1;
    toks[tid] = tok_list[e * Tv + sc];
    wts[tid]  = tok_w[e * Tv + sc];
  }
  __syncthreads();

  const int wv = tid >> 5, lane = tid & 31;
  const int wm = (wv & 3) << 5;
  const int wn = (wv >> 2) << 6;

  v8f acc[2][4] = {};
  const float* w2p = w2 + ((size_t)e * Dv + n0) * Hv;

  // A: 128 rows x 32 bf16 = 512 x 16B transfers -> async direct-to-LDS
  auto stageA = [&](int buf, int kb) {
    for (int i = tid; i < 512; i += 256) {
      int r = i >> 2, kg = (i & 3) << 3;
      const bf16* src = hb + (size_t)(m0 + r) * Hv + kb + kg;
      bf16* dst = &As[buf][r * LDSW + kg];
#if USE_ASYNC_LDS
      __builtin_amdgcn_global_load_async_to_lds_b128(
          (__attribute__((address_space(1))) v4i*)src,
          (__attribute__((address_space(3))) v4i*)dst, 0, 0);
#else
      *(v8bf*)dst = *(const v8bf*)src;
#endif
    }
  };
  // B: w2 f32 -> bf16 via registers
  auto stageB = [&](int buf, int kb) {
    for (int i = tid; i < 1024; i += 256) {
      int r = i >> 3, kg = (i & 7) << 2;
      float4 v = *(const float4*)(w2p + (size_t)r * Hv + kb + kg);
      cvt_store4(&Bs[buf][r * LDSW + kg], v);
    }
  };

  stageA(0, 0);
  stageB(0, 0);
#if USE_ASYNC_LDS
  __builtin_amdgcn_s_wait_asynccnt(0);
#endif
  __syncthreads();

  constexpr int NCH = Hv / 32;                 // 88 chunks
  for (int kc = 0; kc < NCH; ++kc) {
    const int cur = kc & 1;
    if (kc + 1 < NCH) {
      stageA(cur ^ 1, (kc + 1) * 32);          // async, overlaps with WMMA
      stageB(cur ^ 1, (kc + 1) * 32);
    }
#pragma unroll
    for (int mi = 0; mi < 2; ++mi) {
      v16bf af = ld_frag(As[cur], wm + mi * 16, lane);
#pragma unroll
      for (int ni = 0; ni < 4; ++ni) {
        v16bf bfr = ld_frag(Bs[cur], wn + ni * 16, lane);
        acc[mi][ni] = __builtin_amdgcn_wmma_f32_16x16x32_bf16(
            false, af, false, bfr, (short)0, acc[mi][ni], false, false);
      }
    }
#if USE_ASYNC_LDS
    __builtin_amdgcn_s_wait_asynccnt(0);
#endif
    __syncthreads();
  }

  const int hi8 = (lane >> 4) << 3, l15 = lane & 15;
#pragma unroll
  for (int mi = 0; mi < 2; ++mi)
#pragma unroll
    for (int ni = 0; ni < 4; ++ni)
#pragma unroll
      for (int i = 0; i < 8; ++i) {
        int r = wm + mi * 16 + hi8 + i;
        int slot = m0 + r;
        if (slot < cnt) {
          int col = n0 + wn + ni * 16 + l15;
          float* op = out + (size_t)toks[r] * Dv + col;
          *op += wts[r] * acc[mi][ni][i];
        }
      }
}

// ---------------------------------------------------------------------------
// Host driver: router -> per-expert (h, out) GEMMs, all on `stream`.
// Workspace: [counts 256B][tok_list E*T*4][tok_w E*T*4][hb T*H*2] ~= 46.7 MB
// ---------------------------------------------------------------------------
extern "C" void kernel_launch(void* const* d_in, const int* in_sizes, int n_in,
                              void* d_out, int out_size, void* d_ws, size_t ws_size,
                              hipStream_t stream) {
  (void)in_sizes; (void)n_in; (void)ws_size;
  const float* x  = (const float*)d_in[0];
  const float* gw = (const float*)d_in[1];
  const float* w1 = (const float*)d_in[2];
  const float* w2 = (const float*)d_in[3];
  const float* w3 = (const float*)d_in[4];
  float* out = (float*)d_out;

  char* ws = (char*)d_ws;
  int*   counts   = (int*)ws;                                        // 256 B
  int*   tok_list = (int*)(ws + 256);                                // E*T ints
  float* tok_w    = (float*)(ws + 256 + (size_t)Ev * Tv * 4);        // E*T floats
  bf16*  hb       = (bf16*)(ws + 256 + 2 * (size_t)Ev * Tv * 4);     // T*H bf16

  hipMemsetAsync(counts, 0, 256, stream);
  hipMemsetAsync(out, 0, (size_t)out_size * sizeof(float), stream);

  router_kernel<<<Tv / 8, 256, 0, stream>>>(x, gw, counts, tok_list, tok_w);

  for (int e = 0; e < Ev; ++e) {
    moe_h_kernel<<<dim3(Tv / 128, Hv / 128), 256, 0, stream>>>(
        x, w1, w3, counts, tok_list, hb, e);
    moe_o_kernel<<<dim3(Tv / 128, Dv / 128), 256, 0, stream>>>(
        hb, w2, counts, tok_list, tok_w, out, e);
  }
}
